// RetinaNet_6305011990818
// MI455X (gfx1250) — matrix operand order repaired and checked by reading, not compiled, not run on hardware
//
#include <hip/hip_runtime.h>

typedef __attribute__((ext_vector_type(16))) _Float16 v16h;
typedef __attribute__((ext_vector_type(8)))  _Float16 v8h;
typedef __attribute__((ext_vector_type(8)))  float    v8f;
typedef __attribute__((ext_vector_type(4)))  int      v4i;

#define CIN     256
#define KSTEPS  72        // 9 taps * 8 chunks of K=32
#define OUTROWS 13        // 4 bbox rows + 9 class rows
#define PIXH    264       // halo pixel stride in halfs (512B + 16B pad)

#define AS1 __attribute__((address_space(1)))
#define AS3 __attribute__((address_space(3)))

#ifndef __has_builtin
#define __has_builtin(x) 0
#endif
#if __has_builtin(__builtin_amdgcn_global_load_async_to_lds_b128)
#define USE_ASYNC_LDS 1
#else
#define USE_ASYNC_LDS 0
#endif

union ABFrag { v16h v; v8h h[2]; };

__device__ __forceinline__ v8f wmma16(const ABFrag& a, const ABFrag& b, v8f c) {
  return __builtin_amdgcn_wmma_f32_16x16x32_f16(false, a.v, false, b.v,
                                                (short)0, c, false, false);
}

// ---------------- NCHW f32 -> NHWC f16 ----------------
__global__ void __launch_bounds__(256)
k_nchw_to_nhwc(const float* __restrict__ in, _Float16* __restrict__ out,
               int HW, int total /* B*HW*CIN */) {
  int idx = blockIdx.x * 256 + threadIdx.x;
  if (idx >= total) return;
  int c  = idx & (CIN - 1);
  int np = idx >> 8;           // n*HW + p
  int n  = np / HW;
  int p  = np - n * HW;
  out[idx] = (_Float16)in[((size_t)n * CIN + c) * HW + p];
}

// ------- OIHW f32 weights -> packed WMMA B fragments (f16) -------
// wp[((nt*KSTEPS + ks)*32 + lane)*16 + h] = W[k][n]
//   n = nt*16 + (lane&15)
//   k = ks*32 + ((lane&16)>>1) + (h&7) + ((h&8)<<1) ;  k = tap*256 + cin ; tap = ky*3+kx
__global__ void __launch_bounds__(256)
k_pack_w(const float* __restrict__ w, _Float16* __restrict__ wp,
         int Cout, int totalHalf) {
  int idx = blockIdx.x * 256 + threadIdx.x;
  if (idx >= totalHalf) return;
  int h    = idx & 15;
  int lane = (idx >> 4) & 31;
  int ksnt = idx >> 9;
  int ks   = ksnt % KSTEPS;
  int nt   = ksnt / KSTEPS;
  int n    = nt * 16 + (lane & 15);
  int k    = ks * 32 + ((lane & 16) >> 1) + (h & 7) + ((h & 8) << 1);
  int tap  = k >> 8;
  int cin  = k & 255;
  int ky = tap / 3, kx = tap - ky * 3;
  float val = 0.f;
  if (n < Cout) val = w[(((size_t)n * CIN + cin) * 3 + ky) * 3 + kx];
  wp[idx] = (_Float16)val;
}

// ------- 3x3 SAME conv (Cin=256) as implicit GEMM via WMMA f16 -------
// Block tile = 8x8 pixels x 128 outC. 8 waves = 2 M-waves x 4 N-waves;
// each wave computes a 2x2 grid of 16x16 subtiles (32 px x 32 outC) so each
// (A0,A1,B0,B1) fragment quad feeds 4 WMMAs. 10x10 halo staged in LDS via
// async global->LDS copies (zero-filled borders via disjoint ds_stores).
// MODE 0: stem -> f16 NHWC out (Cout=256), +bias, ReLU
// MODE 1: head -> f32 scattered into final [N,13,Atotal] layout
template<int MODE>
__global__ void __launch_bounds__(256)
k_conv3x3(const _Float16* __restrict__ xin, const _Float16* __restrict__ wp,
          const float* __restrict__ bias,
          _Float16* __restrict__ outh, float* __restrict__ outf,
          int H, int W, int tilesX, int Cout,
          int rowBase, int Aoff, int Atotal)
{
  __shared__ __align__(16) _Float16 lds[100 * PIXH];   // 52.8 KB

  const int img = blockIdx.z;
  const int tX  = blockIdx.x % tilesX;
  const int tY  = blockIdx.x / tilesX;
  const int HWl = H * W;
  const _Float16* xim = xin + (size_t)img * HWl * CIN;

  // ---- stage 10x10 halo x 256ch into LDS ----
  for (int i = threadIdx.x; i < 3200; i += 256) {      // 3200 x 16B chunks
    const int pix  = i >> 5;          // 0..99
    const int part = i & 31;          // 0..31 (8 halfs each)
    const int hy = pix / 10;
    const int hx = pix - hy * 10;
    const int gy = tY * 8 - 1 + hy;
    const int gx = tX * 8 - 1 + hx;
    const int ldsOff = pix * PIXH + part * 8;
    const bool valid = (unsigned)gy < (unsigned)H && (unsigned)gx < (unsigned)W;
#if USE_ASYNC_LDS
    if (valid) {
      __builtin_amdgcn_global_load_async_to_lds_b128(
          (AS1 v4i*)(xim + ((size_t)(gy * W + gx)) * CIN + part * 8),
          (AS3 v4i*)&lds[ldsOff], 0, 0);
    } else {
      const v8h z = {};
      *(v8h*)&lds[ldsOff] = z;                         // zero-fill border
    }
#else
    v8h v = {};
    if (valid) v = *(const v8h*)(xim + ((size_t)(gy * W + gx)) * CIN + part * 8);
    *(v8h*)&lds[ldsOff] = v;
#endif
  }
#if USE_ASYNC_LDS
#if __has_builtin(__builtin_amdgcn_s_wait_asynccnt)
  __builtin_amdgcn_s_wait_asynccnt(0);
#else
  asm volatile("s_wait_asynccnt 0" ::: "memory");
#endif
#endif

  const int lane = threadIdx.x & 31;
  const int wave = threadIdx.x >> 5;
  const int mW   = wave >> 2;                 // 0..1 (32-pixel M half)
  const int nW   = wave & 3;                  // 0..3 (32-channel N quarter)
  const int m    = lane & 15;
  const int hiK  = (lane & 16) ? 8 : 0;       // ISA 16-bit A lane K-split
  const int py0  = mW * 4 + (m >> 3);         // pixel row, M-subtile 0
  const int px   = m & 7;
  const int nt0  = blockIdx.y * 8 + nW * 2;
  const int nCol0 = nt0 * 16 + m;
  const _Float16* wb0 = wp + (size_t)nt0 * KSTEPS * 512 + lane * 16;
  const _Float16* wb1 = wb0 + (size_t)KSTEPS * 512;

  v8f acc00 = {}, acc01 = {}, acc10 = {}, acc11 = {};
  __syncthreads();

  // ---- 72-step K loop, fully unrolled, branch-free, 4 WMMA / step ----
#pragma unroll
  for (int tap = 0; tap < 9; ++tap) {
    const int dy = tap / 3;
    const int dx = tap - dy * 3;              // halo coords: (py+dy, px+dx)
    const int a0 = ((py0 + dy) * 10 + (px + dx)) * PIXH + hiK;
    const int a1 = a0 + 2 * 10 * PIXH;        // M-subtile 1 is 2 rows below
#pragma unroll
    for (int kc = 0; kc < 8; ++kc) {
      const int ks = tap * 8 + kc;
      ABFrag A0, A1, B0, B1;
      A0.h[0] = *(const v8h*)&lds[a0 + kc * 32];
      A0.h[1] = *(const v8h*)&lds[a0 + kc * 32 + 16];
      A1.h[0] = *(const v8h*)&lds[a1 + kc * 32];
      A1.h[1] = *(const v8h*)&lds[a1 + kc * 32 + 16];
      const _Float16* p0 = wb0 + (size_t)ks * 512;
      const _Float16* p1 = wb1 + (size_t)ks * 512;
      B0.h[0] = *(const v8h*)p0;  B0.h[1] = *(const v8h*)(p0 + 8);
      B1.h[0] = *(const v8h*)p1;  B1.h[1] = *(const v8h*)(p1 + 8);
      acc00 = wmma16(A0, B0, acc00);
      acc01 = wmma16(A0, B1, acc01);
      acc10 = wmma16(A1, B0, acc10);
      acc11 = wmma16(A1, B1, acc11);
    }
  }

  // ---- epilogue ----
  const int mBase = (lane & 16) ? 8 : 0;      // C/D: hi lanes hold rows 8..15
  const float bv0 = (nCol0 < Cout) ? bias[nCol0] : 0.f;
  const float bv1 = (nCol0 + 16 < Cout) ? bias[nCol0 + 16] : 0.f;
#pragma unroll
  for (int s = 0; s < 2; ++s) {
#pragma unroll
    for (int r = 0; r < 8; ++r) {
      const int mm = mBase + r;
      const int gy = tY * 8 + (mW * 2 + s) * 2 + (mm >> 3);
      const int gx = tX * 8 + (mm & 7);
      if (gy < H && gx < W) {
#pragma unroll
        for (int j = 0; j < 2; ++j) {
          const int nc = nCol0 + j * 16;
          if (nc < Cout) {
            const v8f& a = s ? (j ? acc11 : acc10) : (j ? acc01 : acc00);
            float v = a[r] + (j ? bv1 : bv0);
            if (MODE == 0) {
              v = v > 0.f ? v : 0.f;
              outh[((size_t)img * HWl + gy * W + gx) * CIN + nc] = (_Float16)v;
            } else {
              const int rem2 = gy * W + gx;
              const int a6 = nc / 6;
              const int b6 = nc - a6 * 6;
              outf[((size_t)img * OUTROWS + rowBase + a6) * (size_t)Atotal
                   + (size_t)Aoff + (size_t)b6 * HWl + rem2] = v;
            }
          }
        }
      }
    }
  }
}

extern "C" void kernel_launch(void* const* d_in, const int* in_sizes, int n_in,
                              void* d_out, int out_size, void* d_ws, size_t ws_size,
                              hipStream_t stream) {
  (void)in_sizes; (void)n_in; (void)out_size; (void)ws_size;
  const float* feat[5] = {(const float*)d_in[0], (const float*)d_in[1],
                          (const float*)d_in[2], (const float*)d_in[3],
                          (const float*)d_in[4]};
  const float* reg_stem_w = (const float*)d_in[5];
  const float* reg_stem_b = (const float*)d_in[6];
  const float* reg_head_w = (const float*)d_in[7];
  const float* reg_head_b = (const float*)d_in[8];
  const float* cls_stem_w = (const float*)d_in[9];
  const float* cls_stem_b = (const float*)d_in[10];
  const float* cls_head_w = (const float*)d_in[11];
  const float* cls_head_b = (const float*)d_in[12];
  float* out = (float*)d_out;

  const int HWs[5]   = {64, 32, 16, 8, 4};
  const int Aoffs[5] = {0, 24576, 30720, 32256, 32640};
  const int Atotal   = 32736;

  // ---- workspace carve ----
  char* ws = (char*)d_ws;
  size_t off = 0;
  const size_t stemPkHalf = 16ull * KSTEPS * 512;  // Cout 256 -> 16 N-tiles
  const size_t headPkHalf = 8ull  * KSTEPS * 512;  // heads padded to 128 cols
  _Float16* regStemPk = (_Float16*)(ws + off); off += 20ull * stemPkHalf * 2;
  _Float16* clsStemPk = (_Float16*)(ws + off); off += 20ull * stemPkHalf * 2;
  _Float16* regHeadPk = (_Float16*)(ws + off); off += 5ull * headPkHalf * 2;
  _Float16* clsHeadPk = (_Float16*)(ws + off); off += 5ull * headPkHalf * 2;
  const size_t bufHalf = 8192ull * CIN;            // max level: 2*64*64 pixels
  _Float16* bufIn = (_Float16*)(ws + off); off += bufHalf * 2;
  _Float16* bufA  = (_Float16*)(ws + off); off += bufHalf * 2;
  _Float16* bufB  = (_Float16*)(ws + off); off += bufHalf * 2;

  // ---- pack all weights ----
  {
    const int e = (int)stemPkHalf;
    dim3 g((e + 255) / 256);
    for (int i = 0; i < 20; ++i) {
      k_pack_w<<<g, 256, 0, stream>>>(reg_stem_w + (size_t)i * CIN * CIN * 9,
                                      regStemPk + (size_t)i * stemPkHalf, 256, e);
      k_pack_w<<<g, 256, 0, stream>>>(cls_stem_w + (size_t)i * CIN * CIN * 9,
                                      clsStemPk + (size_t)i * stemPkHalf, 256, e);
    }
    const int eh = (int)headPkHalf;
    dim3 gh((eh + 255) / 256);
    for (int l = 0; l < 5; ++l) {
      k_pack_w<<<gh, 256, 0, stream>>>(reg_head_w + (size_t)l * 24 * CIN * 9,
                                       regHeadPk + (size_t)l * headPkHalf, 24, eh);
      k_pack_w<<<gh, 256, 0, stream>>>(cls_head_w + (size_t)l * 54 * CIN * 9,
                                       clsHeadPk + (size_t)l * headPkHalf, 54, eh);
    }
  }

  // ---- per-level subnets ----
  for (int l = 0; l < 5; ++l) {
    const int HW = HWs[l], HW2 = HW * HW, total = 2 * HW2;
    {
      const int e = total * CIN;
      k_nchw_to_nhwc<<<dim3((e + 255) / 256), 256, 0, stream>>>(feat[l], bufIn, HW2, e);
    }
    const int tilesX = (HW + 7) / 8;
    const int tiles  = tilesX * tilesX;
    const dim3 gs(tiles, 2, 2);               // stems: 256 cols / 128 per block
    _Float16* bufs[2] = {bufA, bufB};

    // reg subnet
    const _Float16* xp = bufIn;
    for (int cv = 0; cv < 4; ++cv) {
      _Float16* yo = bufs[cv & 1];
      k_conv3x3<0><<<gs, 256, 0, stream>>>(xp, regStemPk + (size_t)(l * 4 + cv) * stemPkHalf,
                                           reg_stem_b + (size_t)(l * 4 + cv) * CIN,
                                           yo, nullptr, HW, HW, tilesX, CIN, 0, 0, 0);
      xp = yo;
    }
    k_conv3x3<1><<<dim3(tiles, 1, 2), 256, 0, stream>>>(
        xp, regHeadPk + (size_t)l * headPkHalf, reg_head_b + (size_t)l * 24,
        nullptr, out, HW, HW, tilesX, 24, 0, Aoffs[l], Atotal);

    // cls subnet
    xp = bufIn;
    for (int cv = 0; cv < 4; ++cv) {
      _Float16* yo = bufs[cv & 1];
      k_conv3x3<0><<<gs, 256, 0, stream>>>(xp, clsStemPk + (size_t)(l * 4 + cv) * stemPkHalf,
                                           cls_stem_b + (size_t)(l * 4 + cv) * CIN,
                                           yo, nullptr, HW, HW, tilesX, CIN, 0, 0, 0);
      xp = yo;
    }
    k_conv3x3<1><<<dim3(tiles, 1, 2), 256, 0, stream>>>(
        xp, clsHeadPk + (size_t)l * headPkHalf, cls_head_b + (size_t)l * 54,
        nullptr, out, HW, HW, tilesX, 54, 4, Aoffs[l], Atotal);
  }
}
